// WindowAttention_64364379898115
// MI455X (gfx1250) — matrix operand order repaired and checked by reading, not compiled, not run on hardware
//
#include <hip/hip_runtime.h>

typedef __attribute__((ext_vector_type(16))) _Float16 v16h;
typedef __attribute__((ext_vector_type(8)))  _Float16 v8h;
typedef __attribute__((ext_vector_type(4)))  _Float16 v4h;
typedef __attribute__((ext_vector_type(2)))  _Float16 v2h;
typedef __attribute__((ext_vector_type(8)))  float    v8f;

#define DIM      512
#define HEADS    16
#define HD       32
#define NTOK     64
#define NWIN     2048
#define W_GAIN   0.044194173824159216f   // 1/sqrt(512)
#define QK_SCALE 0.17677669529663687f    // 1/sqrt(32)

// ---- dynamic LDS layout (bytes); all sub-buffers 16B aligned, strides keep
// ---- b128 accesses 16B aligned and dword accesses 4B aligned.
#define NX_OFF   0         // 64 x NX_LD f16  (normalized x)
#define NX_LD    520
#define S_OFF    66560     // 64 x S_LD f32   (attention logits)
#define S_LD     72
#define P_OFF    84992     // 64 x P_LD f16   (softmax probabilities)
#define P_LD     72
#define Q_OFF    94208     // 64 x QK_LD f16  (q head slice)
#define QK_LD    40
#define K_OFF    99328     // 64 x QK_LD f16  (k head slice)
#define VT_OFF   104448    // 32 x VT_LD f16  (v head slice, transposed [d][tok])
#define VT_LD    72
#define OB_OFF   109056    // 64 x OB_LD f16  (full attention output, all heads)
#define OB_LD    520
#define NRM_OFF  175616    // 64 f32 row norms
#define TOK_OFF  175872    // 64 f32 token mask (-100 / 0)
#define MWV_OFF  176128    // 1 f32 mw value
#define SMEM_BYTES 176144

// ---------- fragment loaders (work for LDS or global pointers) ----------

// A fragment: 16x32 f16, row-major source [row][k], stride ld (halves).
// ISA layout: lanes hold M=lane%16; vgpr j holds K pair at
//   K = (j>>2)*16 + (j&3)*2 + (lane/16)*8.
__device__ __forceinline__ v16h load_A16(const _Float16* base, int ld,
                                         int row0, int col0, int lane) {
  const int g = (lane >> 4) & 1, m = lane & 15;
  const _Float16* p = base + (size_t)(row0 + m) * ld + col0;
  v16h r;
#pragma unroll
  for (int j = 0; j < 8; ++j) {
    int k = ((j >> 2) << 4) + ((j & 3) << 1) + (g << 3);
    v2h t = *(const v2h*)(p + k);             // dword load (pair K,K+1)
    r[2 * j]     = t[0];
    r[2 * j + 1] = t[1];
  }
  return r;
}

// B fragment: 32x16 f16 where B(k,n) = src[n][k] (i.e. B^T row-major),
// stride ld (halves). Lane holds N=lane%16, K = (lane/16)*8 + j (j<8),
// 16 + (lane/16)*8 + (j-8) (j>=8) -> two b128.
__device__ __forceinline__ v16h load_B16(const _Float16* base, int ld,
                                         int n0, int k0, int lane) {
  const int g = (lane >> 4) & 1, n = lane & 15;
  const _Float16* p = base + (size_t)(n0 + n) * ld + k0 + (g << 3);
  v8h lo = *(const v8h*)(p);
  v8h hi = *(const v8h*)(p + 16);
  return __builtin_shufflevector(lo, hi, 0, 1, 2, 3, 4, 5, 6, 7,
                                 8, 9, 10, 11, 12, 13, 14, 15);
}

// ---------- weight prep: f32 -> f16 pre-scaled by W_GAIN ----------
__global__ __launch_bounds__(256) void prep_weights_kernel(
    const float* __restrict__ qw, const float* __restrict__ kw,
    const float* __restrict__ vw, const float* __restrict__ pw,
    _Float16* __restrict__ w16) {
  size_t i = (size_t)blockIdx.x * 256 + threadIdx.x;  // 4 * 512*512 total
  size_t mat = i >> 18;
  size_t off = i & ((size_t(1) << 18) - 1);
  const float* src = (mat == 0) ? qw : (mat == 1) ? kw : (mat == 2) ? vw : pw;
  w16[i] = (_Float16)(src[off] * W_GAIN);
}

// ---------- fused window attention ----------
__global__ __launch_bounds__(256) void win_attn_kernel(
    const float* __restrict__ x, const float* __restrict__ mask,
    const float* __restrict__ mwin,
    const float* __restrict__ qb, const float* __restrict__ kb,
    const float* __restrict__ vb, const float* __restrict__ pb,
    const _Float16* __restrict__ w16,   // [4][512][512] f16, pre-scaled
    float* __restrict__ out, float* __restrict__ mwout) {
  extern __shared__ char smem[];
  _Float16* nxb  = (_Float16*)(smem + NX_OFF);
  float*    Sb   = (float*)   (smem + S_OFF);
  _Float16* Pb   = (_Float16*)(smem + P_OFF);
  _Float16* qbuf = (_Float16*)(smem + Q_OFF);
  _Float16* kbuf = (_Float16*)(smem + K_OFF);
  _Float16* vT   = (_Float16*)(smem + VT_OFF);
  _Float16* Ob   = (_Float16*)(smem + OB_OFF);
  float* nrm = (float*)(smem + NRM_OFF);
  float* tok = (float*)(smem + TOK_OFF);
  float* mwv = (float*)(smem + MWV_OFF);

  const int b    = blockIdx.x;
  const int tid  = threadIdx.x;
  const int wave = tid >> 5;
  const int lane = tid & 31;
  const int g    = (lane >> 4) & 1;
  const int ln   = lane & 15;

  // ---- phase 0: row L2-normalize x -> nx (f16, LDS), token mask, mw ----
  {
    const int row = tid >> 2, quad = tid & 3;   // 4 lanes per row
    const float4* xr =
        (const float4*)(x + ((size_t)b * NTOK + row) * DIM + quad * 128);
    float ss = 0.f;
#pragma unroll 8
    for (int i = 0; i < 32; ++i) {
      float4 t = xr[i];
      ss += t.x * t.x + t.y * t.y + t.z * t.z + t.w * t.w;
    }
    ss += __shfl_xor(ss, 1, 32);
    ss += __shfl_xor(ss, 2, 32);
    float nn  = fmaxf(sqrtf(ss), 1e-12f);
    float inv = 1.f / nn;
    if (quad == 0) nrm[row] = nn;
    _Float16* dst = nxb + row * NX_LD + quad * 128;
#pragma unroll 8
    for (int i = 0; i < 32; ++i) {
      float4 t = xr[i];
      v4h pk;
      pk[0] = (_Float16)(t.x * inv);
      pk[1] = (_Float16)(t.y * inv);
      pk[2] = (_Float16)(t.z * inv);
      pk[3] = (_Float16)(t.w * inv);
      *(v4h*)(dst + i * 4) = pk;
    }
    if (tid < 64) {
      float m = mwin[(size_t)b * NTOK + tid];
      tok[tid] = (m == 0.f) ? -100.f : 0.f;
    }
    if (tid == 0) {
      float s = 0.f;
      for (int n = 0; n < NTOK; ++n) s += mwin[(size_t)b * NTOK + n];
      mwv[0] = fminf(fmaxf(s, 0.f), 1.f);
    }
  }
  __syncthreads();

  const v8f vzero = {0.f, 0.f, 0.f, 0.f, 0.f, 0.f, 0.f, 0.f};

  for (int h = 0; h < HEADS; ++h) {
    // ---- 1a: q/k/v head slices: nx[64,512] @ W_h^T[512,32]  ----
#pragma unroll
    for (int i = 0; i < 3; ++i) {
      const int id  = wave * 3 + i;        // 24 tiles over 8 waves
      const int mat = id >> 3;             // 0=q 1=k 2=v (wave-uniform)
      const int t   = id & 7;
      const int mr  = t & 3, nc = t >> 2;
      const _Float16* W = w16 + (size_t)mat * DIM * DIM;
      v8f acc = vzero;
#pragma unroll
      for (int kk = 0; kk < 16; ++kk) {
        v16h a  = load_A16(nxb, NX_LD, mr * 16, kk * 32, lane);
        v16h bf = load_B16(W, DIM, h * HD + nc * 16, kk * 32, lane);
        acc = __builtin_amdgcn_wmma_f32_16x16x32_f16(false, a, false, bf,
                                                     (short)0, acc, false, false);
      }
      const int   o  = h * HD + nc * 16 + ln;
      const float bs = (mat == 0) ? qb[o] : (mat == 1) ? kb[o] : vb[o];
      if (mat == 2) {
        // v row scaled back by ||x||; store transposed [d][tok] contiguously
        v8h pk;
#pragma unroll
        for (int r = 0; r < 8; ++r)
          pk[r] = (_Float16)(acc[r] * nrm[mr * 16 + g * 8 + r] + bs);
        *(v8h*)(vT + (size_t)(nc * 16 + ln) * VT_LD + mr * 16 + g * 8) = pk;
      } else {
        _Float16* dst = (mat == 0) ? qbuf : kbuf;
#pragma unroll
        for (int r = 0; r < 8; ++r)
          dst[(size_t)(mr * 16 + g * 8 + r) * QK_LD + nc * 16 + ln] =
              (_Float16)(acc[r] + bs);
      }
    }
    __syncthreads();

    // ---- 1b: logits S = q k^T * scale + win_mask + token_mask ----
#pragma unroll
    for (int i = 0; i < 2; ++i) {
      const int t  = wave * 2 + i;         // 16 tiles over 8 waves
      const int mr = t & 3, nc = t >> 2;
      v16h a  = load_A16(qbuf, QK_LD, mr * 16, 0, lane);
      v16h bf = load_B16(kbuf, QK_LD, nc * 16, 0, lane);
      v8f acc = __builtin_amdgcn_wmma_f32_16x16x32_f16(false, a, false, bf,
                                                       (short)0, vzero, false, false);
      const int   col  = nc * 16 + ln;
      const float tkm  = tok[col];
      const float* mrow = mask + (size_t)(b & 63) * (64 * 64) + col;
#pragma unroll
      for (int r = 0; r < 8; ++r) {
        int m = mr * 16 + g * 8 + r;
        Sb[(size_t)m * S_LD + col] = acc[r] * QK_SCALE + mrow[m * 64] + tkm;
      }
    }
    __syncthreads();

    // ---- 1c: softmax along rows (4 lanes x 16 cols per row) ----
    {
      const int row = tid >> 2, quad = tid & 3;
      float* sr = Sb + (size_t)row * S_LD + quad * 16;
      float e[16];
      float mx = -3.0e38f;
#pragma unroll
      for (int j = 0; j < 16; ++j) { e[j] = sr[j]; mx = fmaxf(mx, e[j]); }
      mx = fmaxf(mx, __shfl_xor(mx, 1, 32));
      mx = fmaxf(mx, __shfl_xor(mx, 2, 32));
      float sm = 0.f;
#pragma unroll
      for (int j = 0; j < 16; ++j) { e[j] = __expf(e[j] - mx); sm += e[j]; }
      sm += __shfl_xor(sm, 1, 32);
      sm += __shfl_xor(sm, 2, 32);
      const float inv = 1.f / sm;
      _Float16* pr = Pb + (size_t)row * P_LD + quad * 16;
#pragma unroll
      for (int j = 0; j < 16; ++j) pr[j] = (_Float16)(e[j] * inv);
    }
    __syncthreads();

    // ---- 1d: O_h = P[64,64] @ v_h[64,32] -> columns h*32.. of Ob ----
    {
      const int mr = wave & 3, nc = wave >> 2;   // 8 tiles over 8 waves
      v8f acc = vzero;
#pragma unroll
      for (int kk = 0; kk < 2; ++kk) {
        v16h a  = load_A16(Pb, P_LD, mr * 16, kk * 32, lane);
        v16h bf = load_B16(vT, VT_LD, nc * 16, kk * 32, lane);
        acc = __builtin_amdgcn_wmma_f32_16x16x32_f16(false, a, false, bf,
                                                     (short)0, acc, false, false);
      }
#pragma unroll
      for (int r = 0; r < 8; ++r)
        Ob[(size_t)(mr * 16 + g * 8 + r) * OB_LD + h * HD + nc * 16 + ln] =
            (_Float16)acc[r];
    }
    __syncthreads();
  }

  // ---- epilogue: out = Ob[64,512] @ Wp^T[512,512] + pb (K=512 GEMM) ----
  {
    const int mr    = wave & 3;                  // row tile
    const int cbase = (wave >> 2) * 16;          // 16 col tiles per wave
    const _Float16* Wp = w16 + (size_t)3 * DIM * DIM;
    float* op = out + (size_t)b * NTOK * DIM;
#pragma unroll
    for (int grp = 0; grp < 4; ++grp) {          // 4 tiles per group -> 4 accs
      v8f acc0 = vzero, acc1 = vzero, acc2 = vzero, acc3 = vzero;
      const int c0 = (cbase + grp * 4) * 16;
#pragma unroll
      for (int kk = 0; kk < 16; ++kk) {
        if (kk + 1 < 16)                          // warm next K-slice of Wp
          __builtin_prefetch(Wp + (size_t)(c0 + ln) * DIM + (kk + 1) * 32, 0, 1);
        v16h a = load_A16(Ob, OB_LD, mr * 16, kk * 32, lane);
        v16h b0 = load_B16(Wp, DIM, c0,      kk * 32, lane);
        v16h b1 = load_B16(Wp, DIM, c0 + 16, kk * 32, lane);
        v16h b2 = load_B16(Wp, DIM, c0 + 32, kk * 32, lane);
        v16h b3 = load_B16(Wp, DIM, c0 + 48, kk * 32, lane);
        acc0 = __builtin_amdgcn_wmma_f32_16x16x32_f16(false, a, false, b0,
                                                      (short)0, acc0, false, false);
        acc1 = __builtin_amdgcn_wmma_f32_16x16x32_f16(false, a, false, b1,
                                                      (short)0, acc1, false, false);
        acc2 = __builtin_amdgcn_wmma_f32_16x16x32_f16(false, a, false, b2,
                                                      (short)0, acc2, false, false);
        acc3 = __builtin_amdgcn_wmma_f32_16x16x32_f16(false, a, false, b3,
                                                      (short)0, acc3, false, false);
      }
      v8f* accs[4] = {&acc0, &acc1, &acc2, &acc3};
#pragma unroll
      for (int j = 0; j < 4; ++j) {
        const int   o  = c0 + j * 16 + ln;
        const float bs = pb[o];
#pragma unroll
        for (int r = 0; r < 8; ++r) {
          int m = mr * 16 + g * 8 + r;
          op[(size_t)m * DIM + o] = (*accs[j])[r] + bs;
        }
      }
    }
    if (tid < 64) mwout[(size_t)b * NTOK + tid] = mwv[0];
  }
}

extern "C" void kernel_launch(void* const* d_in, const int* in_sizes, int n_in,
                              void* d_out, int out_size, void* d_ws,
                              size_t ws_size, hipStream_t stream) {
  const float* x    = (const float*)d_in[0];
  const float* mask = (const float*)d_in[1];
  const float* mwin = (const float*)d_in[2];
  const float* qw   = (const float*)d_in[3];
  const float* qb   = (const float*)d_in[4];
  const float* kw   = (const float*)d_in[5];
  const float* kb   = (const float*)d_in[6];
  const float* vw   = (const float*)d_in[7];
  const float* vb   = (const float*)d_in[8];
  const float* pw   = (const float*)d_in[9];
  const float* pb   = (const float*)d_in[10];

  float* out   = (float*)d_out;
  float* mwout = out + (size_t)NWIN * NTOK * DIM;
  _Float16* w16 = (_Float16*)d_ws;   // 4 * 512*512 f16 = 2 MB

  prep_weights_kernel<<<4096, 256, 0, stream>>>(qw, kw, vw, pw, w16);

  (void)hipFuncSetAttribute((const void*)win_attn_kernel,
                            hipFuncAttributeMaxDynamicSharedMemorySize,
                            SMEM_BYTES);
  win_attn_kernel<<<NWIN, 256, SMEM_BYTES, stream>>>(
      x, mask, mwin, qb, kb, vb, pb, w16, out, mwout);
}
